// MuleDetectorGNN_7035156430981
// MI455X (gfx1250) — matrix operand order repaired and checked by reading, not compiled, not run on hardware
//
#include <hip/hip_runtime.h>
#include <hip/hip_bf16.h>
#include <stdint.h>

#define NNODES 100000
#define NEDGES 1600000
#define BN_EPS 1e-5f

typedef __attribute__((ext_vector_type(16))) __bf16 v16bf;
typedef __attribute__((ext_vector_type(8)))  float  v8f;
typedef unsigned short u16b;

union Frag16 { v16bf v; uint4 q[2]; };

__device__ __forceinline__ u16b f32_to_bf16(float f) {
  unsigned int u = __float_as_uint(f);
  u += 0x7fffu + ((u >> 16) & 1u);   // round-to-nearest-even
  return (u16b)(u >> 16);
}

// ---------------- graph aggregation ----------------

__global__ void k_deg(const int* __restrict__ dst, float* __restrict__ deg, int E) {
  int e = blockIdx.x * blockDim.x + threadIdx.x;
  if (e < E) atomicAdd(&deg[dst[e]], 1.0f);
}

__global__ void k_scatter(const float* __restrict__ X, const int* __restrict__ src,
                          const int* __restrict__ dst, float* __restrict__ agg,
                          int F, int E) {
  int e = blockIdx.x * blockDim.x + threadIdx.x;
  if (e >= E) return;
  const float* xr = X + (size_t)src[e] * F;
  float*       ar = agg + (size_t)dst[e] * F;
  for (int j = 0; j < F; ++j) atomicAdd(&ar[j], xr[j]);
}

// agg[n, j<F] / max(deg,1)  ->  bf16 [n, Fpad] (zero padded)
__global__ void k_mean_cvt(const float* __restrict__ agg, const float* __restrict__ deg,
                           u16b* __restrict__ out, int F, int Fpad, int total) {
  int t = blockIdx.x * blockDim.x + threadIdx.x;
  if (t >= total) return;
  int n = t / Fpad, j = t - n * Fpad;
  float v = 0.f;
  if (j < F) v = agg[(size_t)n * F + j] / fmaxf(deg[n], 1.0f);
  out[t] = f32_to_bf16(v);
}

// fp32 [n,F] -> bf16 [n,Fpad] zero padded
__global__ void k_cvt_pad(const float* __restrict__ X, u16b* __restrict__ out,
                          int F, int Fpad, int total) {
  int t = blockIdx.x * blockDim.x + threadIdx.x;
  if (t >= total) return;
  int n = t / Fpad, j = t - n * Fpad;
  float v = (j < F) ? X[(size_t)n * F + j] : 0.f;
  out[t] = f32_to_bf16(v);
}

// W [K,O] fp32 row-major -> Wt [O,Kpad] bf16 (transposed, K zero padded)
__global__ void k_prep_w(const float* __restrict__ W, u16b* __restrict__ Wt,
                         int K, int O, int Kpad) {
  int t = blockIdx.x * blockDim.x + threadIdx.x;
  if (t >= O * Kpad) return;
  int o = t / Kpad, kp = t - o * Kpad;
  float v = (kp < K) ? W[(size_t)kp * O + o] : 0.f;
  Wt[t] = f32_to_bf16(v);
}

// ---------------- WMMA GEMM: Y = A1@W1t^T [+ A2@W2t^T] [+bias] [relu] ----------------
// A: [M, Kpad] bf16 row-major.  Wt: [O, Kpad] bf16 (i.e. W transposed).
// grid.x = M/16 ; block = 128 (4 waves) ; grid.y = ceil(O/64); wave w -> cols [.. +16)
__global__ __launch_bounds__(128)
void k_wmma_gemm(const u16b* __restrict__ A1, const u16b* __restrict__ A2,
                 const u16b* __restrict__ W1t, const u16b* __restrict__ W2t,
                 const float* __restrict__ bias, float* __restrict__ Y,
                 int Kpad, int O, int do_relu) {
  int m0 = blockIdx.x << 4;
  int w  = threadIdx.x >> 5;
  int n0 = (blockIdx.y * 4 + w) << 4;
  if (n0 >= O) return;                    // whole-wave exit: EXEC stays all-ones for WMMA
  int lane = threadIdx.x & 31;
  int g = lane >> 4;                      // half-wave group
  int r = lane & 15;

  v8f c = {0.f, 0.f, 0.f, 0.f, 0.f, 0.f, 0.f, 0.f};

  {
    const u16b* ar = A1  + (size_t)(m0 + r) * Kpad + g * 8;
    const u16b* br = W1t + (size_t)(n0 + r) * Kpad + g * 8;
    for (int k0 = 0; k0 < Kpad; k0 += 32) {
      Frag16 a, b;
      a.q[0] = *reinterpret_cast<const uint4*>(ar + k0);        // K = g*8 .. +7
      a.q[1] = *reinterpret_cast<const uint4*>(ar + k0 + 16);   // K = 16+g*8 .. +7
      b.q[0] = *reinterpret_cast<const uint4*>(br + k0);        // K = g*8 .. +7
      b.q[1] = *reinterpret_cast<const uint4*>(br + k0 + 16);   // K = 16+g*8 .. +7
      c = __builtin_amdgcn_wmma_f32_16x16x32_bf16(false, a.v, false, b.v,
                                                  (short)0, c, false, false);
    }
  }
  if (A2 != nullptr) {
    const u16b* ar = A2  + (size_t)(m0 + r) * Kpad + g * 8;
    const u16b* br = W2t + (size_t)(n0 + r) * Kpad + g * 8;
    for (int k0 = 0; k0 < Kpad; k0 += 32) {
      Frag16 a, b;
      a.q[0] = *reinterpret_cast<const uint4*>(ar + k0);
      a.q[1] = *reinterpret_cast<const uint4*>(ar + k0 + 16);
      b.q[0] = *reinterpret_cast<const uint4*>(br + k0);
      b.q[1] = *reinterpret_cast<const uint4*>(br + k0 + 16);
      c = __builtin_amdgcn_wmma_f32_16x16x32_bf16(false, a.v, false, b.v,
                                                  (short)0, c, false, false);
    }
  }

  float bv = bias ? bias[n0 + r] : 0.f;
  for (int i = 0; i < 8; ++i) {           // lanes 0-15: M=i ; lanes 16-31: M=8+i
    int row = m0 + i + g * 8;
    float y = c[i] + bv;
    if (do_relu) y = fmaxf(y, 0.f);
    Y[(size_t)row * O + n0 + r] = y;
  }
}

// H = relu(bn(Y)) + R ; also emit bf16 copy for next layer's GEMMs
__global__ void k_bn_relu_add(const float* __restrict__ Yv, const float* __restrict__ R,
                              const float* __restrict__ gg, const float* __restrict__ bb,
                              const float* __restrict__ rm, const float* __restrict__ rv,
                              float* __restrict__ Hout, u16b* __restrict__ Hbf,
                              int F, int total) {
  int t = blockIdx.x * blockDim.x + threadIdx.x;
  if (t >= total) return;
  int j = t % F;
  float y = (Yv[t] - rm[j]) * rsqrtf(rv[j] + BN_EPS) * gg[j] + bb[j];
  y = fmaxf(y, 0.f) + R[t];
  Hout[t] = y;
  Hbf[t]  = f32_to_bf16(y);
}

// ---------------- scalar heads ----------------

__global__ void k_head_mule(const float* __restrict__ clsh, const float* __restrict__ W,
                            const float* __restrict__ b, float* __restrict__ out, int n_nodes) {
  int n = blockIdx.x * blockDim.x + threadIdx.x;
  if (n >= n_nodes) return;
  const float* h = clsh + (size_t)n * 32;
  float s = b[0];
  for (int i = 0; i < 32; ++i) s += h[i] * W[i];
  out[n] = 1.0f / (1.0f + expf(-s));
}

__global__ void k_head_risk(const float* __restrict__ exph, const float* __restrict__ W,
                            const float* __restrict__ b, float* __restrict__ out, int n_nodes) {
  int n = blockIdx.x * blockDim.x + threadIdx.x;
  if (n >= n_nodes) return;
  const float* h = exph + (size_t)n * 16;
  float l[5];
  float m = -1e30f;
  for (int j = 0; j < 5; ++j) {
    float s = b[j];
    for (int i = 0; i < 16; ++i) s += h[i] * W[i * 5 + j];
    l[j] = s; m = fmaxf(m, s);
  }
  float den = 0.f;
  for (int j = 0; j < 5; ++j) { l[j] = expf(l[j] - m); den += l[j]; }
  float inv = 1.0f / den;
  for (int j = 0; j < 5; ++j) out[(size_t)n * 5 + j] = l[j] * inv;
}

// ---------------- host launch ----------------

static inline int cdiv(int a, int b) { return (a + b - 1) / b; }

extern "C" void kernel_launch(void* const* d_in, const int* in_sizes, int n_in,
                              void* d_out, int out_size, void* d_ws, size_t ws_size,
                              hipStream_t stream) {
  (void)in_sizes; (void)n_in; (void)out_size; (void)ws_size;
  const int N = NNODES, E = NEDGES;

  const float* x    = (const float*)d_in[0];
  const int*   ei   = (const int*)d_in[1];
  const int*   src  = ei;
  const int*   dst  = ei + E;
  const float* c1_Wl=(const float*)d_in[2],  *c1_bl=(const float*)d_in[3],  *c1_Wr=(const float*)d_in[4];
  const float* c2_Wl=(const float*)d_in[5],  *c2_bl=(const float*)d_in[6],  *c2_Wr=(const float*)d_in[7];
  const float* c3_Wl=(const float*)d_in[8],  *c3_bl=(const float*)d_in[9],  *c3_Wr=(const float*)d_in[10];
  const float* bn1_g=(const float*)d_in[11], *bn1_b=(const float*)d_in[12], *bn1_rm=(const float*)d_in[13], *bn1_rv=(const float*)d_in[14];
  const float* bn2_g=(const float*)d_in[15], *bn2_b=(const float*)d_in[16], *bn2_rm=(const float*)d_in[17], *bn2_rv=(const float*)d_in[18];
  const float* bn3_g=(const float*)d_in[19], *bn3_b=(const float*)d_in[20], *bn3_rm=(const float*)d_in[21], *bn3_rv=(const float*)d_in[22];
  const float* r1_W=(const float*)d_in[23],  *r1_b=(const float*)d_in[24];
  const float* r2_W=(const float*)d_in[25],  *r2_b=(const float*)d_in[26];
  const float* r3_W=(const float*)d_in[27],  *r3_b=(const float*)d_in[28];
  const float* cl1_W=(const float*)d_in[29], *cl1_b=(const float*)d_in[30];
  const float* cl2_W=(const float*)d_in[31], *cl2_b=(const float*)d_in[32];
  const float* ex1_W=(const float*)d_in[33], *ex1_b=(const float*)d_in[34];
  const float* ex2_W=(const float*)d_in[35], *ex2_b=(const float*)d_in[36];

  // workspace bump allocator
  char* ws = (char*)d_ws;
  size_t off = 0;
  auto alloc = [&](size_t bytes) -> char* {
    char* p = ws + off;
    off = (off + bytes + 255) & ~(size_t)255;
    return p;
  };
  float* deg    = (float*)alloc((size_t)N * 4);
  float* aggf   = (float*)alloc((size_t)N * 256 * 4);   // agg fp32, reused as R later
  float* Ybuf   = (float*)alloc((size_t)N * 256 * 4);
  u16b*  aggbf  = (u16b*) alloc((size_t)N * 256 * 2);
  u16b*  selfbf = (u16b*) alloc((size_t)N * 256 * 2);   // xbf -> h1bf -> h2bf -> h3bf
  float* h1     = (float*)alloc((size_t)N * 128 * 4);
  float* h2     = (float*)alloc((size_t)N * 256 * 4);
  float* clsh   = (float*)alloc((size_t)N * 32 * 4);
  float* exph   = (float*)alloc((size_t)N * 16 * 4);
  u16b* wl1t = (u16b*)alloc(128 * 32 * 2);
  u16b* wr1t = (u16b*)alloc(128 * 32 * 2);
  u16b* rr1t = (u16b*)alloc(128 * 32 * 2);
  u16b* wl2t = (u16b*)alloc(256 * 128 * 2);
  u16b* wr2t = (u16b*)alloc(256 * 128 * 2);
  u16b* rr2t = (u16b*)alloc(256 * 128 * 2);
  u16b* wl3t = (u16b*)alloc(64 * 256 * 2);
  u16b* wr3t = (u16b*)alloc(64 * 256 * 2);
  u16b* rr3t = (u16b*)alloc(64 * 256 * 2);
  u16b* cl1t = (u16b*)alloc(32 * 64 * 2);
  u16b* ex1t = (u16b*)alloc(16 * 64 * 2);
  float* R = aggf;    // residual GEMM output aliases agg buffer (agg dead by then)

  const int T = 256;
  // degrees (once; same graph all layers)
  hipMemsetAsync(deg, 0, (size_t)N * 4, stream);
  k_deg<<<cdiv(E, T), T, 0, stream>>>(dst, deg, E);

  // weight prep: bf16 transposed, K padded
  k_prep_w<<<cdiv(128 * 32, T), T, 0, stream>>>(c1_Wl, wl1t, 10, 128, 32);
  k_prep_w<<<cdiv(128 * 32, T), T, 0, stream>>>(c1_Wr, wr1t, 10, 128, 32);
  k_prep_w<<<cdiv(128 * 32, T), T, 0, stream>>>(r1_W,  rr1t, 10, 128, 32);
  k_prep_w<<<cdiv(256 * 128, T), T, 0, stream>>>(c2_Wl, wl2t, 128, 256, 128);
  k_prep_w<<<cdiv(256 * 128, T), T, 0, stream>>>(c2_Wr, wr2t, 128, 256, 128);
  k_prep_w<<<cdiv(256 * 128, T), T, 0, stream>>>(r2_W,  rr2t, 128, 256, 128);
  k_prep_w<<<cdiv(64 * 256, T), T, 0, stream>>>(c3_Wl, wl3t, 256, 64, 256);
  k_prep_w<<<cdiv(64 * 256, T), T, 0, stream>>>(c3_Wr, wr3t, 256, 64, 256);
  k_prep_w<<<cdiv(64 * 256, T), T, 0, stream>>>(r3_W,  rr3t, 256, 64, 256);
  k_prep_w<<<cdiv(32 * 64, T), T, 0, stream>>>(cl1_W, cl1t, 64, 32, 64);
  k_prep_w<<<cdiv(16 * 64, T), T, 0, stream>>>(ex1_W, ex1t, 64, 16, 64);

  // x -> bf16 padded (10 -> 32)
  k_cvt_pad<<<cdiv(N * 32, T), T, 0, stream>>>(x, selfbf, 10, 32, N * 32);

  // ---- layer 1 (K=10 pad 32, O=128) ----
  hipMemsetAsync(aggf, 0, (size_t)N * 10 * 4, stream);
  k_scatter<<<cdiv(E, T), T, 0, stream>>>(x, src, dst, aggf, 10, E);
  k_mean_cvt<<<cdiv(N * 32, T), T, 0, stream>>>(aggf, deg, aggbf, 10, 32, N * 32);
  k_wmma_gemm<<<dim3(N / 16, cdiv(128, 64)), 128, 0, stream>>>(
      aggbf, selfbf, wl1t, wr1t, c1_bl, Ybuf, 32, 128, 0);
  k_wmma_gemm<<<dim3(N / 16, cdiv(128, 64)), 128, 0, stream>>>(
      selfbf, nullptr, rr1t, nullptr, r1_b, R, 32, 128, 0);
  k_bn_relu_add<<<cdiv(N * 128, T), T, 0, stream>>>(
      Ybuf, R, bn1_g, bn1_b, bn1_rm, bn1_rv, h1, selfbf, 128, N * 128);

  // ---- layer 2 (K=128, O=256) ----
  hipMemsetAsync(aggf, 0, (size_t)N * 128 * 4, stream);
  k_scatter<<<cdiv(E, T), T, 0, stream>>>(h1, src, dst, aggf, 128, E);
  k_mean_cvt<<<cdiv(N * 128, T), T, 0, stream>>>(aggf, deg, aggbf, 128, 128, N * 128);
  k_wmma_gemm<<<dim3(N / 16, cdiv(256, 64)), 128, 0, stream>>>(
      aggbf, selfbf, wl2t, wr2t, c2_bl, Ybuf, 128, 256, 0);
  k_wmma_gemm<<<dim3(N / 16, cdiv(256, 64)), 128, 0, stream>>>(
      selfbf, nullptr, rr2t, nullptr, r2_b, R, 128, 256, 0);
  k_bn_relu_add<<<cdiv(N * 256, T), T, 0, stream>>>(
      Ybuf, R, bn2_g, bn2_b, bn2_rm, bn2_rv, h2, selfbf, 256, N * 256);

  // ---- layer 3 (K=256, O=64) ----
  hipMemsetAsync(aggf, 0, (size_t)N * 256 * 4, stream);
  k_scatter<<<cdiv(E, T), T, 0, stream>>>(h2, src, dst, aggf, 256, E);
  k_mean_cvt<<<cdiv(N * 256, T), T, 0, stream>>>(aggf, deg, aggbf, 256, 256, N * 256);
  k_wmma_gemm<<<dim3(N / 16, 1), 128, 0, stream>>>(
      aggbf, selfbf, wl3t, wr3t, c3_bl, Ybuf, 256, 64, 0);
  k_wmma_gemm<<<dim3(N / 16, 1), 128, 0, stream>>>(
      selfbf, nullptr, rr3t, nullptr, r3_b, R, 256, 64, 0);
  float* out_f = (float*)d_out;
  float* h3    = out_f + N;                        // h3 lives directly in d_out
  k_bn_relu_add<<<cdiv(N * 64, T), T, 0, stream>>>(
      Ybuf, R, bn3_g, bn3_b, bn3_rm, bn3_rv, h3, selfbf, 64, N * 64);

  // ---- heads ----
  k_wmma_gemm<<<dim3(N / 16, 1), 128, 0, stream>>>(
      selfbf, nullptr, cl1t, nullptr, cl1_b, clsh, 64, 32, 1);
  k_wmma_gemm<<<dim3(N / 16, 1), 128, 0, stream>>>(
      selfbf, nullptr, ex1t, nullptr, ex1_b, exph, 64, 16, 1);
  k_head_mule<<<cdiv(N, T), T, 0, stream>>>(clsh, cl2_W, cl2_b, out_f, N);
  k_head_risk<<<cdiv(N, T), T, 0, stream>>>(exph, ex2_W, ex2_b, out_f + N + (size_t)N * 64, N);
}
